// BoundaryLoss_27608049779436
// MI455X (gfx1250) — matrix operand order repaired and checked by reading, not compiled, not run on hardware
//
#include <hip/hip_runtime.h>
#include <hip/hip_bf16.h>

#define NK     150
#define NB     512
#define FD     768
#define NCOL   576          // 512 ood columns + 64 fused per-label sample columns
#define TRILN  294528       // 768*767/2
#define MAXS   64
#define BETA_C 0.3f
#define KSTEP  64
#define LDA    68           // padded LDS stride in bf16 elements (from 64)
#define NSTEPS (FD / KSTEP) // 12

typedef __attribute__((ext_vector_type(16))) __bf16 v16bf;
typedef __attribute__((ext_vector_type(8)))  float  v8f;

union Frag { v16bf h; unsigned int u[8]; };

__device__ __forceinline__ unsigned int bf16_1(float f) {
  unsigned int u = __float_as_uint(f);
  u += 0x7FFFu + ((u >> 16) & 1u);          // round-to-nearest-even
  return u >> 16;
}
__device__ __forceinline__ unsigned int bf16_pk(float lo, float hi) {
  unsigned int ul = __float_as_uint(lo); ul += 0x7FFFu + ((ul >> 16) & 1u);
  unsigned int uh = __float_as_uint(hi); uh += 0x7FFFu + ((uh >> 16) & 1u);
  return (ul >> 16) | (uh & 0xFFFF0000u);
}

// ---------------- init: zero sumsq + accumulators + counts ----------------
__global__ void bl_init(float* __restrict__ sumsq_accum, int* __restrict__ counts) {
  int i = blockIdx.x * 256 + threadIdx.x;
  if (i < NK * NCOL + 4) sumsq_accum[i] = 0.0f;
  if (i < NK)            counts[i] = 0;
}

// ---------------- bucket samples by label ----------------
__global__ void bl_bucket(const int* __restrict__ labels, int* __restrict__ counts,
                          int* __restrict__ lists) {
  int b = blockIdx.x * 256 + threadIdx.x;
  if (b < NB) {
    int lab = labels[b];
    if (lab >= 0 && lab < NK) {
      int slot = atomicAdd(&counts[lab], 1);
      if (slot < MAXS) lists[lab * MAXS + slot] = b;
    }
  }
}

// ---- A-tile fill: R_k[i0..i0+127, j0..j0+63] -> bf16 LDS, triangle-specialized ----
__device__ __forceinline__ void fill_A(unsigned short* __restrict__ As,
    const float* __restrict__ Lm, const float* __restrict__ Um,
    const float* __restrict__ Dd, unsigned kT, unsigned kF,
    int i0, int j0, int tid) {
  if (i0 >= j0 + KSTEP) {
    // pure lower: L row gi is contiguous along j -> 4 consecutive loads/thread
    #pragma unroll
    for (int it = 0; it < 8; ++it) {
      int q = it * 256 + tid;              // 2048 quads
      int i = q >> 4;                      // 0..127
      int c = (q & 15) * 4;                // 0..60
      unsigned gi = (unsigned)(i0 + i);
      unsigned base = kT + (gi * (gi - 1u)) / 2u + (unsigned)(j0 + c);
      float a0 = Lm[base + 0], a1 = Lm[base + 1];
      float a2 = Lm[base + 2], a3 = Lm[base + 3];
      uint2 st; st.x = bf16_pk(a0, a1); st.y = bf16_pk(a2, a3);
      *reinterpret_cast<uint2*>(&As[i * LDA + c]) = st;
    }
  } else if (j0 >= i0 + 128) {
    // pure upper: U column gj is contiguous along i -> 4 consecutive loads/thread
    #pragma unroll
    for (int it = 0; it < 8; ++it) {
      int j = it * 8 + (tid >> 5);         // 0..63
      int i = (tid & 31) * 4;              // 0..124
      unsigned gj = (unsigned)(j0 + j);
      unsigned base = kT + (gj * (gj - 1u)) / 2u + (unsigned)(i0 + i);
      As[(i + 0) * LDA + j] = (unsigned short)bf16_1(Um[base + 0]);
      As[(i + 1) * LDA + j] = (unsigned short)bf16_1(Um[base + 1]);
      As[(i + 2) * LDA + j] = (unsigned short)bf16_1(Um[base + 2]);
      As[(i + 3) * LDA + j] = (unsigned short)bf16_1(Um[base + 3]);
    }
  } else {
    // diagonal-band tile: per-element select (12 of 72 tiles per k)
    for (int e = tid; e < 128 * KSTEP; e += 256) {
      int i = e >> 6, j = e & 63;
      unsigned gi = (unsigned)(i0 + i), gj = (unsigned)(j0 + j);
      float v;
      if (gi > gj)      v = Lm[kT + (gi * (gi - 1u)) / 2u + gj];
      else if (gi < gj) v = Um[kT + (gj * (gj - 1u)) / 2u + gi];
      else              v = Dd[kF + gi];
      As[i * LDA + j] = (unsigned short)bf16_1(v);
    }
  }
}

// ---- B-tile fill: columns 0..511 = ood - c_k ; 512.. = pooled - c_k (fused part 1) ----
__device__ __forceinline__ void fill_B(unsigned short* __restrict__ Bs,
    const float* __restrict__ ood, const float* __restrict__ pooled,
    const float* __restrict__ ck, const int* __restrict__ lists,
    int cnt, int k, int n0, int j0, int tid) {
  #pragma unroll
  for (int it = 0; it < 4; ++it) {
    int q = it * 256 + tid;                // 1024 quads
    int n = q >> 4;                        // 0..63
    int c = (q & 15) * 4;                  // 0..60
    int gn = n0 + n;
    unsigned gj = (unsigned)(j0 + c);
    float a0 = 0.f, a1 = 0.f, a2 = 0.f, a3 = 0.f;
    const float* src = nullptr;
    if (gn < NB) {
      src = ood + (unsigned)gn * FD;
    } else {
      int s = gn - NB;
      if (s < cnt) src = pooled + (unsigned)lists[k * MAXS + s] * FD;
    }
    if (src) {
      a0 = src[gj + 0] - ck[gj + 0];
      a1 = src[gj + 1] - ck[gj + 1];
      a2 = src[gj + 2] - ck[gj + 2];
      a3 = src[gj + 3] - ck[gj + 3];
    }
    uint2 st; st.x = bf16_pk(a0, a1); st.y = bf16_pk(a2, a3);
    *reinterpret_cast<uint2*>(&Bs[n * LDA + c]) = st;
  }
}

// ---- one K-step (K=64): 2 chunks x 4 WMMAs, 64 rows x 16 cols per wave ----
__device__ __forceinline__ void compute_step(const unsigned short* __restrict__ As,
    const unsigned short* __restrict__ Bs, v8f* acc,
    int lane, int nsub, int mgrp) {
  #pragma unroll
  for (int cc = 0; cc < KSTEP; cc += 32) {
    Frag bf;
    {
      int n  = nsub * 16 + (lane & 15);
      int kb = cc + ((lane < 16) ? 0 : 8);
      #pragma unroll
      for (int v = 0; v < 8; ++v)
        bf.u[v] = *reinterpret_cast<const unsigned int*>(&Bs[n * LDA + kb + 2 * v + ((v >= 4) ? 8 : 0)]);
    }
    #pragma unroll
    for (int t = 0; t < 4; ++t) {
      Frag af;
      int row = mgrp * 64 + t * 16 + (lane & 15);
      int klo = cc + ((lane < 16) ? 0 : 8);
      int khi = cc + ((lane < 16) ? 16 : 24);
      #pragma unroll
      for (int v = 0; v < 4; ++v)
        af.u[v] = *reinterpret_cast<const unsigned int*>(&As[row * LDA + klo + 2 * v]);
      #pragma unroll
      for (int v = 0; v < 4; ++v)
        af.u[4 + v] = *reinterpret_cast<const unsigned int*>(&As[row * LDA + khi + 2 * v]);
      acc[t] = __builtin_amdgcn_wmma_f32_16x16x32_bf16(
          false, af.h, false, bf.h, (short)0, acc[t], false, false);
    }
  }
}

// ---------------- fused R-build + batched GEMM + row-norm ----------------
// grid = (9 n-tiles, 6 m-tiles, 150 k), block = 256 (8 waves), double-buffered LDS
__global__ __launch_bounds__(256)
void bl_gemm(const float* __restrict__ Lm, const float* __restrict__ Um,
             const float* __restrict__ Dd, const float* __restrict__ cent,
             const float* __restrict__ ood, const float* __restrict__ pooled,
             const int* __restrict__ counts, const int* __restrict__ lists,
             float* __restrict__ sumsq) {
  __shared__ unsigned short As[2][128 * LDA];
  __shared__ unsigned short Bs[2][64 * LDA];

  const int k    = blockIdx.z;
  const int i0   = blockIdx.y * 128;
  const int n0   = blockIdx.x * 64;
  const int tid  = threadIdx.x;
  const int lane = tid & 31;
  const int wave = tid >> 5;
  const int nsub = wave & 3;       // 16-column group within the 64-col tile
  const int mgrp = wave >> 2;      // rows [mgrp*64, mgrp*64+64)
  const int cnt  = counts[k];

  const unsigned kT = (unsigned)k * TRILN;   // fits 32-bit: max ~44.2M
  const unsigned kF = (unsigned)k * FD;
  const float* ck = cent + kF;

  const v8f vzero = {0.f, 0.f, 0.f, 0.f, 0.f, 0.f, 0.f, 0.f};
  v8f acc[4];
  acc[0] = vzero; acc[1] = vzero; acc[2] = vzero; acc[3] = vzero;

  // prologue: fill buffer 0
  fill_A(As[0], Lm, Um, Dd, kT, kF, i0, 0, tid);
  fill_B(Bs[0], ood, pooled, ck, lists, cnt, k, n0, 0, tid);

  for (int s = 0; s < NSTEPS; ++s) {
    __syncthreads();                      // buffer (s&1) is ready
    const int cur = s & 1;
    compute_step(As[cur], Bs[cur], acc, lane, nsub, mgrp);
    if (s + 1 < NSTEPS) {                 // prefetch next K-step into other buffer
      const int j0n = (s + 1) * KSTEP;
      fill_A(As[cur ^ 1], Lm, Um, Dd, kT, kF, i0, j0n, tid);
      fill_B(Bs[cur ^ 1], ood, pooled, ck, lists, cnt, k, n0, j0n, tid);
    }
  }

  // ---- fused row-norm epilogue: sum rx^2 over this wave's 64 M-rows ----
  float s = 0.0f;
  #pragma unroll
  for (int t = 0; t < 4; ++t) {
    #pragma unroll
    for (int v = 0; v < 8; ++v) s += acc[t][v] * acc[t][v];
  }
  s += __shfl_xor(s, 16, 32);   // fold lane-halves (rows v and v+8 share a column)
  if (lane < 16)
    atomicAdd(&sumsq[(unsigned)k * NCOL + (unsigned)(n0 + nsub * 16 + lane)], s);
}

// ---------------- per-(k,n) loss terms + global accumulation ----------------
__global__ void bl_reduce(const float* __restrict__ sumsq, const float* __restrict__ delta,
                          const int* __restrict__ counts, float* __restrict__ accum) {
  __shared__ float sh[4];      // posSum, negSum, posNum, negNum
  if (threadIdx.x < 4) sh[threadIdx.x] = 0.0f;
  __syncthreads();
  int idx = blockIdx.x * 256 + threadIdx.x;
  if (idx < NK * NCOL) {
    int k = idx / NCOL, n = idx - k * NCOL;
    float d    = delta[k];
    float dist = sqrtf(sumsq[idx]);
    if (n < NB) {
      float pl = (d > dist) ? (d - dist + BETA_C) : (BETA_C * __expf(d - dist));
      atomicAdd(&sh[1], pl);
    } else {
      int s = n - NB;
      if (s < counts[k]) {
        float euc = dist;
        if (euc >= d) atomicAdd(&sh[0], euc - d);   // pos_loss (out_mask)
        if (euc > d)  atomicAdd(&sh[2], 1.0f);      // pos_num
        if (euc < d)  atomicAdd(&sh[3], 1.0f);      // neg_num
      }
    }
  }
  __syncthreads();
  if (threadIdx.x < 4) {
    float v = sh[threadIdx.x];
    if (v != 0.0f) atomicAdd(&accum[threadIdx.x], v);
  }
}

// ---------------- finalize 5 scalars ----------------
__global__ void bl_final(const float* __restrict__ accum, float* __restrict__ out) {
  float pm = accum[0] / (float)NB;
  float nm = accum[1] / (float)NB;
  out[0] = pm;
  out[1] = nm;
  out[2] = accum[2];
  out[3] = accum[3];
  out[4] = pm + nm;
}

extern "C" void kernel_launch(void* const* d_in, const int* in_sizes, int n_in,
                              void* d_out, int out_size, void* d_ws, size_t ws_size,
                              hipStream_t stream) {
  const float* pooled = (const float*)d_in[0];
  const float* cent   = (const float*)d_in[1];
  const float* delta  = (const float*)d_in[2];
  const float* Lm     = (const float*)d_in[3];
  const float* Um     = (const float*)d_in[4];
  const float* Dd     = (const float*)d_in[5];
  const float* ood    = (const float*)d_in[6];
  const int*   labels = (const int*)d_in[7];
  float* out = (float*)d_out;

  // workspace layout
  float* sumsq  = (float*)d_ws;                // NK*NCOL floats
  float* accum  = sumsq + NK * NCOL;           // 4 floats
  int*   counts = (int*)(accum + 4);           // NK ints
  int*   lists  = counts + NK;                 // NK*MAXS ints

  bl_init<<<(NK * NCOL + 4 + 255) / 256, 256, 0, stream>>>(sumsq, counts);
  bl_bucket<<<(NB + 255) / 256, 256, 0, stream>>>(labels, counts, lists);

  dim3 grid(NCOL / 64, FD / 128, NK);          // (9, 6, 150)
  bl_gemm<<<grid, 256, 0, stream>>>(Lm, Um, Dd, cent, ood, pooled, counts, lists, sumsq);

  bl_reduce<<<(NK * NCOL + 255) / 256, 256, 0, stream>>>(sumsq, delta, counts, accum);
  bl_final<<<1, 1, 0, stream>>>(accum, out);
}
